// BiLSTM_CRF_60498909331714
// MI455X (gfx1250) — matrix operand order repaired and checked by reading, not compiled, not run on hardware
//
#include <hip/hip_runtime.h>
#include <hip/hip_bf16.h>

// ---------------- problem constants ----------------
#define BB   32
#define TT   1024
#define EE   256
#define HDD  256
#define G4   1024      // 4*HD
#define HH   512
#define KK   48
#define M1   (BB*TT)   // 32768 rows (time-major: row = t*B + b)
#define APAD 264       // padded LDS row stride (halfs) to spread banks

typedef __attribute__((ext_vector_type(16))) _Float16 v16h;
typedef __attribute__((ext_vector_type(8)))  _Float16 v8h;
typedef __attribute__((ext_vector_type(8)))  float    v8f;

// ---------------- prep kernels ----------------
__global__ void k_f2h(const float* __restrict__ in, _Float16* __restrict__ out, int n) {
    int i = blockIdx.x * 256 + threadIdx.x;
    if (i < n) out[i] = (_Float16)in[i];
}

// gather embedding -> time-major f16 activations: xh[t*B+b][e]
__global__ void k_gather(const int* __restrict__ sent, const float* __restrict__ emb,
                         _Float16* __restrict__ xh) {
    int row = blockIdx.x;          // 0..M1-1
    int e   = threadIdx.x;         // 0..255
    int t = row >> 5, b = row & 31;
    int idx = sent[b * TT + t];
    xh[(long long)row * EE + e] = (_Float16)emb[(long long)idx * EE + e];
}

// ---------------- GEMM1: input gates, both directions ----------------
// Output in C-fragment layout: GinF[dir][mt(2048)][nt(64)][lane(32)][8] (f16)
// acc element v of lane l corresponds to (row mt*16 + (l<16?0:8)+v, col nt*16 + l%16).
__global__ void k_gemm_gates(const _Float16* __restrict__ xh,
                             const _Float16* __restrict__ Wf,
                             const _Float16* __restrict__ Wb,
                             const float* __restrict__ bihf, const float* __restrict__ bhhf,
                             const float* __restrict__ bihb, const float* __restrict__ bhhb,
                             _Float16* __restrict__ GinF) {
    __shared__ _Float16 xt[16 * APAD];   // shared A tile (8.25 KB)

    int tid  = threadIdx.x;
    int wave = tid >> 5;
    int lane = tid & 31;
    long long tile0 = (long long)blockIdx.x * 8;          // 8 tiles per block, same mt
    int dir = (int)(tile0 >> 17);                         // 131072 tiles per dir
    int rem = (int)(tile0 & 131071);
    int mt  = rem >> 6;                                   // shared by whole block
    int nt  = (rem & 63) + wave;                          // per-wave N tile
    const _Float16* W  = dir ? Wb : Wf;
    const float*    b1 = dir ? bihb : bihf;
    const float*    b2 = dir ? bhhb : bhhf;

    // cooperative copy of A tile (16 rows x 256 halfs) into padded LDS
    {
        int r  = tid >> 4;          // 0..15
        int cc = (tid & 15) * 16;   // 0..240 step 16
        *(v16h*)&xt[r * APAD + cc] = *(const v16h*)(xh + (long long)(mt * 16 + r) * EE + cc);
    }
    __syncthreads();

    int nrow = nt * 16 + (lane & 15);
    int aoff = (lane < 16) ? 0 : 16;
    int boff = (lane < 16) ? 0 : 16;
    const _Float16* bBase = W + (long long)nrow * EE + boff;
    const _Float16* aLds  = &xt[(lane & 15) * APAD + aoff];

    v8f acc = {};
    for (int kc = 0; kc < 8; ++kc) {
        v8h alo = *(const v8h*)(aLds + kc * 32);
        v8h ahi = *(const v8h*)(aLds + kc * 32 + 8);
        v16h a;
        for (int i = 0; i < 8; ++i) { a[i] = alo[i]; a[i + 8] = ahi[i]; }
        v16h bfrag = *(const v16h*)(bBase + kc * 32);
        acc = __builtin_amdgcn_wmma_f32_16x16x32_f16(false, a, false, bfrag,
                                                     (short)0, acc, false, false);
    }
    float bias = b1[nrow] + b2[nrow];
    v8h og;
    for (int v = 0; v < 8; ++v) og[v] = (_Float16)(acc[v] + bias);
    long long fidx = ((((long long)dir * 2048 + mt) * 64 + nt) * 32 + lane) * 8;
    *(v8h*)(GinF + fidx) = og;
}

// ---------------- recurrent LSTM (one WG per direction) ----------------
__global__ void __launch_bounds__(1024)
k_lstm_rec(const _Float16* __restrict__ GinF,
           const _Float16* __restrict__ Whf,
           const _Float16* __restrict__ Whb,
           _Float16* __restrict__ hseq /* [2][T][B][HD] f16 */) {
    int dir = blockIdx.x;
    const _Float16* Wh  = dir ? Whb : Whf;
    const _Float16* gin = GinF + (long long)dir * 2048 * 64 * 32 * 8;
    _Float16*       hs  = hseq + (long long)dir * M1 * HDD;

    __shared__ _Float16 hfrag[2 * 8 * 32 * 16];   // h in WMMA A-fragment layout (16 KB)
    __shared__ float    gates[BB * G4];           // 128 KB

    int tid  = threadIdx.x;
    int wave = tid >> 5, lane = tid & 31;

    for (int i = tid; i < 2 * 8 * 32 * 16; i += 1024) hfrag[i] = (_Float16)0.f;

    int eb  = tid >> 5;             // batch this thread owns in elementwise phase
    int ek0 = (tid & 31) * 8;       // first of 8 hidden channels
    float c[8];
    for (int j = 0; j < 8; ++j) c[j] = 0.f;

    // tile assignment: waves 0-15 -> mt=0, waves 16-31 -> mt=1; 4 N-tiles per wave
    int mt    = wave >> 4;
    int ntb   = (wave & 15) * 4;
    int nrow0 = ntb * 16 + (lane & 15);
    int boff  = (lane < 16) ? 0 : 16;
    int mbase = mt * 16 + ((lane < 16) ? 0 : 8);

    __syncthreads();

    for (int s = 0; s < TT; ++s) {
        int t = dir ? (TT - 1 - s) : s;
        v8f acc[4];
        // init accumulators from fragment-layout Gin: one 16B load per tile
        for (int q = 0; q < 4; ++q) {
            long long fidx = (((long long)(t * 2 + mt) * 64 + (ntb + q)) * 32 + lane) * 8;
            v8h g = *(const v8h*)(gin + fidx);
            for (int v = 0; v < 8; ++v) acc[q][v] = (float)g[v];
        }
        for (int kc = 0; kc < 8; ++kc) {
            v16h a = *(const v16h*)&hfrag[((mt * 8 + kc) * 32 + lane) * 16];
            for (int q = 0; q < 4; ++q) {
                int n = nrow0 + q * 16;
                v16h bfrag = *(const v16h*)(Wh + (long long)n * HDD + kc * 32 + boff);
                acc[q] = __builtin_amdgcn_wmma_f32_16x16x32_f16(false, a, false, bfrag,
                                                                (short)0, acc[q], false, false);
            }
        }
        for (int q = 0; q < 4; ++q) {
            int n = nrow0 + q * 16;
            for (int v = 0; v < 8; ++v)
                gates[(mbase + v) * G4 + n] = acc[q][v];
        }
        __syncthreads();

        for (int j = 0; j < 8; ++j) {
            int k = ek0 + j;
            float ig = gates[eb * G4 + k];
            float fg = gates[eb * G4 + 256 + k];
            float gg = gates[eb * G4 + 512 + k];
            float og = gates[eb * G4 + 768 + k];
            float si = 1.f / (1.f + expf(-ig));
            float sf = 1.f / (1.f + expf(-fg));
            float so = 1.f / (1.f + expf(-og));
            c[j] = sf * c[j] + si * tanhf(gg);
            float h = so * tanhf(c[j]);
            _Float16 hh = (_Float16)h;
            hs[((long long)t * BB + eb) * HDD + k] = hh;
            // scatter into A-fragment layout for next step
            int kcc = k >> 5;  int r = k & 31;
            int vhi = (r >= 16) ? 4 : 0;  int r2 = r & 15;
            int lx  = (eb & 15) + ((r >= 16) ? 16 : 0);
            int mtx = eb >> 4;
            int r3  = r2 & 7;
            int e   = r2;
            hfrag[((mtx * 8 + kcc) * 32 + lx) * 16 + e] = hh;
        }
        __syncthreads();
    }
}

// ---------------- GEMM2: output projection -> feats[b][t][48] ----------------
__global__ void k_gemm_out(const _Float16* __restrict__ hseq,
                           const _Float16* __restrict__ Wo,
                           const float* __restrict__ bo,
                           float* __restrict__ feats) {
    int wave = threadIdx.x >> 5, lane = threadIdx.x & 31;
    int tile = blockIdx.x * 8 + wave;   // 2048 * 3 tiles
    int mt = tile / 3, nt = tile % 3;
    int mrow = mt * 16 + (lane & 15);
    int nrow = nt * 16 + (lane & 15);   // < 48
    int aoff = (lane < 16) ? 0 : 16;
    int boff = (lane < 16) ? 0 : 16;

    v8f acc = {};
    for (int kc = 0; kc < 16; ++kc) {
        int k0 = kc * 32;
        int d  = k0 >> 8;               // direction half of concat
        int kk = k0 & 255;
        const _Float16* ab = hseq + ((long long)d * M1 + mrow) * HDD + kk + aoff;
        v8h alo = *(const v8h*)ab;
        v8h ahi = *(const v8h*)(ab + 8);
        v16h a;
        for (int i = 0; i < 8; ++i) { a[i] = alo[i]; a[i + 8] = ahi[i]; }
        v16h bfrag = *(const v16h*)(Wo + (long long)nrow * HH + k0 + boff);
        acc = __builtin_amdgcn_wmma_f32_16x16x32_f16(false, a, false, bfrag,
                                                     (short)0, acc, false, false);
    }
    float bias = bo[nrow];
    int mb = mt * 16 + ((lane < 16) ? 0 : 8);
    for (int v = 0; v < 8; ++v) {
        int r = mb + v;  int t = r >> 5;  int b = r & 31;
        feats[((long long)b * TT + t) * KK + nrow] = acc[v] + bias;
    }
}

// ---------------- Viterbi decode (one block per batch row) ----------------
__global__ void k_viterbi(const float* __restrict__ feats,
                          const float* __restrict__ trans,
                          unsigned char* __restrict__ bptr,
                          float* __restrict__ out) {
    int b = blockIdx.x;
    int j = threadIdx.x;   // 64 threads
    __shared__ float tr[KK * KK];
    __shared__ float fv[KK];
    __shared__ float fvn[KK];
    __shared__ unsigned char tags[TT];

    for (int i = j; i < KK * KK; i += 64) tr[i] = trans[i];
    if (j < KK) fv[j] = (j == KK - 1) ? 0.f : -10000.f;
    __syncthreads();

    const float* fb = feats + (long long)b * TT * KK;
    unsigned char* bp = bptr + (long long)b * TT * KK;

    for (int t = 0; t < TT; ++t) {
        if (j < KK) {
            float best = -3.4e38f;  int bi = 0;
            for (int i = 0; i < KK; ++i) {
                float v = fv[i] + tr[j * KK + i];
                if (v > best) { best = v; bi = i; }
            }
            fvn[j] = best + fb[t * KK + j];
            bp[t * KK + j] = (unsigned char)bi;
        }
        __syncthreads();
        if (j < KK) fv[j] = fvn[j];
        __syncthreads();
    }

    if (j == 0) {
        int best = 0; float bv = fv[0];
        for (int i = 1; i < KK; ++i) if (fv[i] > bv) { bv = fv[i]; best = i; }
        tags[TT - 1] = (unsigned char)best;
        for (int t = TT - 1; t >= 1; --t) tags[t - 1] = bp[t * KK + tags[t]];
        float score = fb[(TT - 1) * KK + tags[0]];
        for (int t = 0; t < TT - 1; ++t) score += tr[tags[t] * KK + tags[t + 1]];
        for (int t = 1; t < TT; ++t)     score += fb[t * KK + tags[t]];
        out[b] = score;
        for (int t = 0; t < TT; ++t) out[32 + b * TT + t] = (float)tags[t];
    }
}

// ---------------- launch ----------------
extern "C" void kernel_launch(void* const* d_in, const int* in_sizes, int n_in,
                              void* d_out, int out_size, void* d_ws, size_t ws_size,
                              hipStream_t stream) {
    const int*   sent  = (const int*)  d_in[0];
    const float* emb   = (const float*)d_in[1];
    const float* Wihf  = (const float*)d_in[2];
    const float* Whhf  = (const float*)d_in[3];
    const float* bihf  = (const float*)d_in[4];
    const float* bhhf  = (const float*)d_in[5];
    const float* Wihb  = (const float*)d_in[6];
    const float* Whhb  = (const float*)d_in[7];
    const float* bihb  = (const float*)d_in[8];
    const float* bhhb  = (const float*)d_in[9];
    const float* Wout  = (const float*)d_in[10];
    const float* bout  = (const float*)d_in[11];
    const float* trans = (const float*)d_in[12];
    float* out = (float*)d_out;

    // workspace carve-up (byte offsets, all >=256B aligned)
    char* ws = (char*)d_ws;
    size_t off = 0;
    _Float16* xh     = (_Float16*)(ws + off); off += (size_t)M1 * EE * 2;          // 16 MB
    _Float16* Wihf_h = (_Float16*)(ws + off); off += (size_t)G4 * EE * 2;
    _Float16* Whhf_h = (_Float16*)(ws + off); off += (size_t)G4 * HDD * 2;
    _Float16* Wihb_h = (_Float16*)(ws + off); off += (size_t)G4 * EE * 2;
    _Float16* Whhb_h = (_Float16*)(ws + off); off += (size_t)G4 * HDD * 2;
    _Float16* Wout_h = (_Float16*)(ws + off); off += 65536;
    _Float16* GinF   = (_Float16*)(ws + off); off += (size_t)2 * M1 * G4 * 2;      // 128 MB
    _Float16* hseq   = (_Float16*)(ws + off); off += (size_t)2 * M1 * HDD * 2;     // 32 MB
    float*    feats  = (float*)   (ws + off); off += (size_t)BB * TT * KK * 4;     // 6 MB
    unsigned char* bptr = (unsigned char*)(ws + off); off += (size_t)BB * TT * KK; // 1.5 MB

    // 1) weight conversion f32 -> f16
    k_f2h<<<(G4 * EE + 255) / 256, 256, 0, stream>>>(Wihf, Wihf_h, G4 * EE);
    k_f2h<<<(G4 * HDD + 255) / 256, 256, 0, stream>>>(Whhf, Whhf_h, G4 * HDD);
    k_f2h<<<(G4 * EE + 255) / 256, 256, 0, stream>>>(Wihb, Wihb_h, G4 * EE);
    k_f2h<<<(G4 * HDD + 255) / 256, 256, 0, stream>>>(Whhb, Whhb_h, G4 * HDD);
    k_f2h<<<(KK * HH + 255) / 256, 256, 0, stream>>>(Wout, Wout_h, KK * HH);

    // 2) embedding gather (time-major, f16)
    k_gather<<<M1, 256, 0, stream>>>(sent, emb, xh);

    // 3) input-gate GEMM (both directions), WMMA, LDS-shared A tile
    k_gemm_gates<<<32768, 256, 0, stream>>>(xh, Wihf_h, Wihb_h,
                                            bihf, bhhf, bihb, bhhb, GinF);

    // 4) recurrent LSTM: one workgroup per direction, WMMA per step
    k_lstm_rec<<<2, 1024, 0, stream>>>(GinF, Whhf_h, Whhb_h, hseq);

    // 5) output projection, WMMA
    k_gemm_out<<<768, 256, 0, stream>>>(hseq, Wout_h, bout, feats);

    // 6) Viterbi decode + score
    k_viterbi<<<BB, 64, 0, stream>>>(feats, trans, bptr, out);
}
